// APP_30227979829323
// MI455X (gfx1250) — matrix operand order repaired and checked by reading, not compiled, not run on hardware
//
#include <hip/hip_runtime.h>
#include <hip/hip_bf16.h>
#include <math.h>

#define B_  4
#define C_  1024
#define D_  1024
#define H_  16
#define DH_ 64

typedef __attribute__((ext_vector_type(16))) _Float16 v16h;
typedef __attribute__((ext_vector_type(8)))  float    v8f;

// ---------------------------------------------------------------------------
// CDNA5 async global->LDS copy (ISA ch.10 / 15.18, tracked by ASYNCcnt).
// 16 contiguous bytes per lane; LDS byte address passed in a VGPR.
// ---------------------------------------------------------------------------
__device__ __forceinline__ void async_copy_b128(void* lds, const void* gptr)
{
    const unsigned ldsOff = (unsigned)(unsigned long long)lds;   // low 32 bits = LDS offset
    asm volatile("global_load_async_to_lds_b128 %0, %1, off"
                 :: "v"(ldsOff), "v"(gptr)
                 : "memory");
}
__device__ __forceinline__ void async_wait0()
{
    asm volatile("s_wait_asynccnt 0x0" ::: "memory");
}

// ---------------------------------------------------------------------------
// WMMA f16 16x16x32 fragment helpers (wave32; contiguous k order chosen by this code, same for A and B)
// A (16x32): lane(half,r): row=r; element j <- K=16*half+j (contiguous 16 per lane)
// B (32x16): lane(half,r): col=r; half j <- K=16*half+j (contiguous 16 per lane)
// C/D f32:   element i <-> (M = i + 8*half, N = r)
// ---------------------------------------------------------------------------
__device__ inline v16h frag_a(const _Float16* base, int stride, int row, int k0, int half)
{
    const _Float16* p = base + row * stride + k0 + 16 * half;
    v16h a;
#pragma unroll
    for (int j = 0; j < 8; ++j) a[j] = p[j];
#pragma unroll
    for (int j = 0; j < 8; ++j) a[8 + j] = p[8 + j];
    return a;
}

__device__ inline v16h frag_b(const _Float16* p)   // p already at +16*half, 16 contiguous
{
    v16h b;
#pragma unroll
    for (int j = 0; j < 16; ++j) b[j] = p[j];
    return b;
}

// ---------------------------------------------------------------------------
// LayerNorm (optional) + f32 -> f16 conversion; one block per 1024-row
// ---------------------------------------------------------------------------
__global__ __launch_bounds__(256) void ln_to_f16(const float* __restrict__ x,
                                                 const float* __restrict__ g,
                                                 const float* __restrict__ bta,
                                                 _Float16* __restrict__ out,
                                                 int applyLN)
{
    __shared__ float red[256];
    const float* xr = x + (size_t)blockIdx.x * D_;
    _Float16* orow  = out + (size_t)blockIdx.x * D_;
    const int t = threadIdx.x;
    float v0[4];
    float s = 0.f;
#pragma unroll
    for (int j = 0; j < 4; ++j) { v0[j] = xr[t + 256 * j]; s += v0[j]; }
    red[t] = s; __syncthreads();
    for (int off = 128; off > 0; off >>= 1) { if (t < off) red[t] += red[t + off]; __syncthreads(); }
    const float mean = red[0] * (1.f / D_);
    __syncthreads();
    float s2 = 0.f;
#pragma unroll
    for (int j = 0; j < 4; ++j) { float d = v0[j] - mean; s2 += d * d; }
    red[t] = s2; __syncthreads();
    for (int off = 128; off > 0; off >>= 1) { if (t < off) red[t] += red[t + off]; __syncthreads(); }
    const float rstd = rsqrtf(red[0] * (1.f / D_) + 1e-5f);
#pragma unroll
    for (int j = 0; j < 4; ++j) {
        const int c = t + 256 * j;
        const float y = applyLN ? ((v0[j] - mean) * rstd * g[c] + bta[c]) : v0[j];
        orow[c] = (_Float16)y;
    }
}

// ---------------------------------------------------------------------------
// Weight convert f32 -> f16 WITH transpose (outT[n][k] = in[k][n]); LDS-tiled
// so both global sides stay coalesced. One-time cost, makes every GEMM B-tile
// staging contiguous (b128 LDS traffic instead of scalar transposed stores).
// ---------------------------------------------------------------------------
__global__ __launch_bounds__(256) void cvtT_f16(const float* __restrict__ in,
                                                _Float16* __restrict__ outT)
{
    __shared__ _Float16 tile[64 * 72];
    const int t = threadIdx.x;
    const int k0 = blockIdx.x * 64, n0 = blockIdx.y * 64;
    {
        const int kr = t >> 2, cg = (t & 3) * 16;
        const float* src = in + (size_t)(k0 + kr) * D_ + n0 + cg;
#pragma unroll
        for (int j = 0; j < 16; ++j) tile[kr * 72 + cg + j] = (_Float16)src[j];
    }
    __syncthreads();
    {
        const int nr = t >> 2, cg = (t & 3) * 16;
        _Float16* dst = outT + (size_t)(n0 + nr) * D_ + k0 + cg;
#pragma unroll
        for (int j = 0; j < 16; ++j) dst[j] = tile[(cg + j) * 72 + nr];
    }
}

// ---------------------------------------------------------------------------
// Tiled WMMA GEMM: Out = A[f16,MxK] * B^T[f16,NxK] + bias  (B pre-transposed)
// Block tile 64x64, 8 waves (4 M x 2 N), K-step 32.
// Tile staging uses CDNA5 async global->LDS b128 copies (ASYNCcnt).
// Output modes: Out16 row-major f16; Out32 row-major f32;
//               OutT per-head transposed f16 [M/1024, N/64, 64, 1024]
//               (contiguous per-lane stores; feeds attn@V B-fragments).
// ---------------------------------------------------------------------------
__global__ __launch_bounds__(256) void gemm_wmma(const _Float16* __restrict__ A,
                                                 const _Float16* __restrict__ BT,
                                                 const float* __restrict__ bias,
                                                 _Float16* __restrict__ Out16,
                                                 _Float16* __restrict__ OutT,
                                                 float* __restrict__ Out32,
                                                 int M, int N, int K)
{
    __shared__ __align__(16) _Float16 As [64 * 40];
    __shared__ __align__(16) _Float16 BsT[64 * 40];   // [n][k]
    const int t = threadIdx.x;
    const int m0 = blockIdx.x * 64, n0 = blockIdx.y * 64;
    const int wave = t >> 5, lane = t & 31;
    const int wm = (wave & 3) * 16, wn = (wave >> 2) * 32;
    const int half = lane >> 4, r = lane & 15;
    const int row = t >> 2, cg = (t & 3) * 8;          // staging: 16B per thread
    v8f c0 = {}, c1 = {};
    for (int k0 = 0; k0 < K; k0 += 32) {
        async_copy_b128(As  + row * 40 + cg, A  + (size_t)(m0 + row) * K + k0 + cg);
        async_copy_b128(BsT + row * 40 + cg, BT + (size_t)(n0 + row) * K + k0 + cg);
        async_wait0();
        __syncthreads();
        v16h a  = frag_a(As, 40, wm + r, 0, half);
        v16h b0 = frag_b(BsT + (wn + r) * 40 + 16 * half);
        v16h b1 = frag_b(BsT + (wn + 16 + r) * 40 + 16 * half);
        c0 = __builtin_amdgcn_wmma_f32_16x16x32_f16(false, a, false, b0, (short)0, c0, false, false);
        c1 = __builtin_amdgcn_wmma_f32_16x16x32_f16(false, a, false, b1, (short)0, c1, false, false);
        __syncthreads();
    }
#pragma unroll
    for (int i = 0; i < 8; ++i) {
        const int gm = m0 + wm + half * 8 + i;
        const int gn = n0 + wn + r;
        const float o0 = c0[i] + bias[gn];
        const float o1 = c1[i] + bias[gn + 16];
        if (Out16) {
            Out16[(size_t)gm * N + gn]      = (_Float16)o0;
            Out16[(size_t)gm * N + gn + 16] = (_Float16)o1;
        }
        if (OutT) {   // [(gm>>10), gn>>6, gn&63, gm&1023]; contiguous in i per lane
            const size_t rb = (size_t)(gm >> 10) * H_;
            OutT[((rb + (gn >> 6)) * DH_ + (gn & 63)) * (size_t)C_ + (gm & 1023)]               = (_Float16)o0;
            OutT[((rb + ((gn + 16) >> 6)) * DH_ + ((gn + 16) & 63)) * (size_t)C_ + (gm & 1023)] = (_Float16)o1;
        }
        if (Out32) {
            Out32[(size_t)gm * N + gn]      = o0;
            Out32[(size_t)gm * N + gn + 16] = o1;
        }
    }
}

// ---------------------------------------------------------------------------
// Fused attention: per (b,h), each block owns 16 query rows x full 1024 keys.
//   scores (WMMA, K=64) -> exact GELU * Dh^-0.5 -> row softmax (LDS reduce)
//   -> attn written ONCE non-temporally (final output, never re-read)
//   -> attn kept f16 in LDS -> attn @ V (WMMA, K=1024 split across wave halves)
// K and V^T fragments stream straight from global (L2-hot, 128 KB/head).
// ---------------------------------------------------------------------------
__global__ __launch_bounds__(256) void fused_attn(const _Float16* __restrict__ Qp,
                                                  const _Float16* __restrict__ Kp,
                                                  const _Float16* __restrict__ VpT,
                                                  float* __restrict__ attn,
                                                  _Float16* __restrict__ OutH)
{
    __shared__ __align__(16) _Float16 Qs[16 * 72];
    __shared__ __align__(16) _Float16 Ps[16 * 1032];
    __shared__ float red [16][128];
    __shared__ float red2[16][8];
    __shared__ float rowmax[16];
    __shared__ float rowsum[16];
    __shared__ float pc[4][256];

    const int t = threadIdx.x;
    const int wave = t >> 5, lane = t & 31;
    const int half = lane >> 4, r = lane & 15;
    const int bh = blockIdx.z, b = bh >> 4, h = bh & 15;
    const int mb = blockIdx.x * 16;
    const size_t qkbase = (size_t)b * C_ * D_ + (size_t)h * DH_;

    // stage 16x64 Q tile (async b128: 16 rows x 4 chunks = 64 lanes' worth)
    if (t < 64) {
        const int qr = t >> 2, cq = (t & 3) * 16;
        async_copy_b128(Qs + qr * 72 + cq,
                        Qp + qkbase + (size_t)(mb + qr) * D_ + cq);
    }
    async_wait0();
    __syncthreads();
    const v16h a0 = frag_a(Qs, 72, r, 0,  half);
    const v16h a1 = frag_a(Qs, 72, r, 32, half);

    // ---- scores: 8 x (16-col tiles) per wave, K=64 ----
    v8f c[8];
#pragma unroll
    for (int tt = 0; tt < 8; ++tt) {
        const int n = wave * 128 + tt * 16 + r;
        const _Float16* kb = Kp + qkbase + (size_t)n * D_ + 16 * half;
        const v16h b0 = frag_b(kb);
        const v16h b1 = frag_b(kb + 32);
        v8f acc = {};
        acc = __builtin_amdgcn_wmma_f32_16x16x32_f16(false, a0, false, b0, (short)0, acc, false, false);
        acc = __builtin_amdgcn_wmma_f32_16x16x32_f16(false, a1, false, b1, (short)0, acc, false, false);
#pragma unroll
        for (int i = 0; i < 8; ++i) {   // exact GELU then Dh^-0.5 scale
            const float s = acc[i];
            acc[i] = 0.5f * s * (1.0f + erff(s * 0.70710678f)) * 0.125f;
        }
        c[tt] = acc;
    }

    // ---- row max (16 rows x 128 partials) ----
#pragma unroll
    for (int i = 0; i < 8; ++i) {
        float m = c[0][i];
#pragma unroll
        for (int tt = 1; tt < 8; ++tt) m = fmaxf(m, c[tt][i]);
        red[8 * half + i][wave * 16 + r] = m;
    }
    __syncthreads();
    if (t < 128) {
        const int ro = t >> 3, sg = t & 7;
        float m = red[ro][sg * 16];
#pragma unroll
        for (int j = 1; j < 16; ++j) m = fmaxf(m, red[ro][sg * 16 + j]);
        red2[ro][sg] = m;
    }
    __syncthreads();
    if (t < 16) {
        float m = red2[t][0];
#pragma unroll
        for (int j = 1; j < 8; ++j) m = fmaxf(m, red2[t][j]);
        rowmax[t] = m;
    }
    __syncthreads();

    // ---- exp + row sum ----
#pragma unroll
    for (int i = 0; i < 8; ++i) {
        const int ro = 8 * half + i;
        const float mx = rowmax[ro];
        float s = 0.f;
#pragma unroll
        for (int tt = 0; tt < 8; ++tt) {
            const float e = __expf(c[tt][i] - mx);
            c[tt][i] = e; s += e;
        }
        red[ro][wave * 16 + r] = s;
    }
    __syncthreads();
    if (t < 128) {
        const int ro = t >> 3, sg = t & 7;
        float s = 0.f;
#pragma unroll
        for (int j = 0; j < 16; ++j) s += red[ro][sg * 16 + j];
        red2[ro][sg] = s;
    }
    __syncthreads();
    if (t < 16) {
        float s = 0.f;
#pragma unroll
        for (int j = 0; j < 8; ++j) s += red2[t][j];
        rowsum[t] = s;
    }
    __syncthreads();

    // ---- normalize: stream attn out (non-temporal), keep f16 copy in LDS ----
    float* arow = attn + (size_t)bh * C_ * C_;
#pragma unroll
    for (int i = 0; i < 8; ++i) {
        const int ro = 8 * half + i;
        const float inv = 1.0f / rowsum[ro];
#pragma unroll
        for (int tt = 0; tt < 8; ++tt) {
            const int col = wave * 128 + tt * 16 + r;
            const float p = c[tt][i] * inv;
            __builtin_nontemporal_store(p, &arow[(size_t)(mb + ro) * C_ + col]);
            Ps[ro * 1032 + col] = (_Float16)p;
        }
    }
    __syncthreads();

    // ---- attn @ V: 16x64 out; waves: col tile = wave&3, K-half = wave>>2 ----
    const int ct = wave & 3;
    const int kh = wave >> 2;
    const _Float16* vhead = VpT + (size_t)bh * DH_ * C_;
    v8f o = {};
    for (int k0 = kh * 512; k0 < kh * 512 + 512; k0 += 32) {
        const v16h a  = frag_a(Ps, 1032, r, k0, half);
        const v16h bb = frag_b(vhead + (size_t)(ct * 16 + r) * C_ + k0 + 16 * half);
        o = __builtin_amdgcn_wmma_f32_16x16x32_f16(false, a, false, bb, (short)0, o, false, false);
    }
    if (kh == 1) {
#pragma unroll
        for (int i = 0; i < 8; ++i) pc[ct][lane * 8 + i] = o[i];
    }
    __syncthreads();
    if (kh == 0) {
#pragma unroll
        for (int i = 0; i < 8; ++i) {
            const float val = o[i] + pc[ct][lane * 8 + i];
            const int gm = mb + 8 * half + i;
            const int gn = ct * 16 + r;
            OutH[((size_t)b * C_ + gm) * D_ + h * DH_ + gn] = (_Float16)val;
        }
    }
}

// ---------------------------------------------------------------------------
// L2 normalize rows (1e-12 clamp), f32 -> f32 final output
// ---------------------------------------------------------------------------
__global__ __launch_bounds__(256) void l2norm_rows(const float* __restrict__ x,
                                                   float* __restrict__ out)
{
    __shared__ float red[256];
    const float* xr = x + (size_t)blockIdx.x * D_;
    float* orow = out + (size_t)blockIdx.x * D_;
    const int t = threadIdx.x;
    float v0[4];
    float ss = 0.f;
#pragma unroll
    for (int j = 0; j < 4; ++j) { v0[j] = xr[t + 256 * j]; ss += v0[j] * v0[j]; }
    red[t] = ss; __syncthreads();
    for (int off = 128; off > 0; off >>= 1) { if (t < off) red[t] += red[t + off]; __syncthreads(); }
    const float inv = 1.0f / fmaxf(sqrtf(red[0]), 1e-12f);
#pragma unroll
    for (int j = 0; j < 4; ++j) orow[t + 256 * j] = v0[j] * inv;
}

// ---------------------------------------------------------------------------
extern "C" void kernel_launch(void* const* d_in, const int* in_sizes, int n_in,
                              void* d_out, int out_size, void* d_ws, size_t ws_size,
                              hipStream_t stream)
{
    (void)in_sizes; (void)n_in; (void)out_size; (void)ws_size;
    const float* q      = (const float*)d_in[0];
    const float* k      = (const float*)d_in[1];
    const float* v      = (const float*)d_in[2];
    const float* ln_k_g = (const float*)d_in[3];
    const float* ln_k_b = (const float*)d_in[4];
    const float* ln_v_g = (const float*)d_in[5];
    const float* ln_v_b = (const float*)d_in[6];
    const float* Wq = (const float*)d_in[7];  const float* bq = (const float*)d_in[8];
    const float* Wk = (const float*)d_in[9];  const float* bk = (const float*)d_in[10];
    const float* Wv = (const float*)d_in[11]; const float* bv = (const float*)d_in[12];
    const float* Wo = (const float*)d_in[13]; const float* bo = (const float*)d_in[14];

    float* out  = (float*)d_out;                    // [B,C,D]
    float* attn = out + (size_t)B_ * C_ * D_;       // [B,H,C,C]

    const size_t BCD = (size_t)B_ * C_ * D_;        // 4M elements
    const size_t DD  = (size_t)D_ * D_;             // 1M elements
    _Float16* qh16   = (_Float16*)d_ws;
    _Float16* kln16  = qh16  + BCD;
    _Float16* vln16  = kln16 + BCD;
    _Float16* wqT16  = vln16 + BCD;                 // W^T [out][in]
    _Float16* wkT16  = wqT16 + DD;
    _Float16* wvT16  = wkT16 + DD;
    _Float16* woT16  = wvT16 + DD;
    _Float16* qp16   = woT16 + DD;
    _Float16* kp16   = qp16 + BCD;
    _Float16* vpT16  = kp16 + BCD;                  // [B,H,64,C]
    _Float16* oh16   = vpT16 + BCD;                 // merged heads
    float*    proj32 = (float*)(oh16 + BCD);

    const int RC = B_ * C_;                         // 4096 rows

    // Phase 0: LN + f16 conversion (q NOT normalized, per the source quirk);
    //          weights converted AND transposed once.
    ln_to_f16<<<RC, 256, 0, stream>>>(q, ln_k_g, ln_k_b, qh16, 0);
    ln_to_f16<<<RC, 256, 0, stream>>>(k, ln_k_g, ln_k_b, kln16, 1);
    ln_to_f16<<<RC, 256, 0, stream>>>(v, ln_v_g, ln_v_b, vln16, 1);
    dim3 gt(16, 16);
    cvtT_f16<<<gt, 256, 0, stream>>>(Wq, wqT16);
    cvtT_f16<<<gt, 256, 0, stream>>>(Wk, wkT16);
    cvtT_f16<<<gt, 256, 0, stream>>>(Wv, wvT16);
    cvtT_f16<<<gt, 256, 0, stream>>>(Wo, woT16);

    // Phase 1: projections (V stored per-head transposed for attn@V B-frags)
    dim3 gp(RC / 64, D_ / 64);
    gemm_wmma<<<gp, 256, 0, stream>>>(qh16,  wqT16, bq, qp16, nullptr, nullptr, RC, D_, D_);
    gemm_wmma<<<gp, 256, 0, stream>>>(kln16, wkT16, bk, kp16, nullptr, nullptr, RC, D_, D_);
    gemm_wmma<<<gp, 256, 0, stream>>>(vln16, wvT16, bv, nullptr, vpT16, nullptr, RC, D_, D_);

    // Phase 2: fused scores+GELU+softmax+attn@V (attn written exactly once)
    dim3 gf(C_ / 16, 1, B_ * H_);
    fused_attn<<<gf, 256, 0, stream>>>(qp16, kp16, vpT16, attn, oh16);

    // Phase 3: output projection + L2 normalize
    gemm_wmma<<<gp, 256, 0, stream>>>(oh16, woT16, bo, nullptr, nullptr, proj32, RC, D_, D_);
    l2norm_rows<<<RC, 256, 0, stream>>>(proj32, out);
}
